// ODEGRUEncoder_29875792511589
// MI455X (gfx1250) — compile-verified, matrix operand order checked by reading, not run on hardware
//
#include <hip/hip_runtime.h>
#include <math.h>

// ODE-GRU encoder for MI455X (gfx1250, wave32, WMMA + TDM).
// Persistent cooperative kernel; per-phase 64x64 C-tiles; operands staged into
// LDS by the Tensor Data Mover (tensor_load_to_lds + s_wait_tensorcnt), with
// D#-descriptor padding giving a bank-conflict-friendly LDS row stride of K+8.
// All GEMMs are v_wmma_f32_16x16x32_bf16 with f32 accumulate.

typedef __bf16 bf16;
typedef __attribute__((ext_vector_type(16))) __bf16 bf16x16;
typedef __attribute__((ext_vector_type(8)))  float  f32x8;
typedef __attribute__((ext_vector_type(4)))  unsigned int u32x4;
typedef __attribute__((ext_vector_type(8)))  int i32x8;
typedef __attribute__((ext_vector_type(4)))  int i32x4;

#define NBLK 128   // persistent blocks (one 64x64 C-tile each for H-wide GEMMs)
#define NTHR 256   // 8 wave32s
#define B_   1024
#define T_   128
#define I_   128
#define H_   512
#define H3_  1536
#define L_   64
#define NSUB 4

#define PADE 8                  // LDS pad: 8 bf16 = 4 DWORDs per row
#define SMEM_ELEMS (64 * (H_ + PADE))        // one 64-row K<=512 panel
#define SMEM_BYTES (2 * SMEM_ELEMS * 2)      // A panel + B panel, bf16

#if defined(__has_builtin)
#  if __has_builtin(__builtin_amdgcn_tensor_load_to_lds) && \
      __has_builtin(__builtin_amdgcn_s_wait_tensorcnt)
#    define HAS_TDM 1
#  endif
#endif
#ifndef HAS_TDM
#  define HAS_TDM 0
#endif

// ---------------------------------------------------------------- grid sync
__device__ __forceinline__ void grid_sync(unsigned* bar) {
  __threadfence();          // every wave flushes its stores to device scope
  __syncthreads();
  if (threadIdx.x == 0) {
    volatile unsigned* vb = (volatile unsigned*)bar;
    unsigned gen = vb[1];
    if (atomicAdd(&bar[0], 1u) == NBLK - 1) {
      bar[0] = 0;
      __threadfence();
      atomicAdd(&bar[1], 1u);            // release next generation
    } else {
      while (vb[1] == gen) __builtin_amdgcn_s_sleep(1);
    }
  }
  __syncthreads();
}

// ------------------------------------------------------------- TDM staging
__device__ __forceinline__ unsigned lds_offset_of(const void* p) {
  // LDS aperture: flat addr[31:0] is the LDS byte offset (ISA 10.2)
  return (unsigned)(unsigned long long)p;
}

#if HAS_TDM
// 2-D tile load: 64 rows x K bf16 from row-major global (leading dim ld
// elements) into LDS at lds_off, with 16B padding appended to every row
// (pad_interval = one row's DWORDs, pad_amount = 4 DWORDs) -> stride K+8.
__device__ __forceinline__ void tdm_load_panel(const bf16* src, int ld,
                                               unsigned lds_off, int K) {
  unsigned long long ga = (unsigned long long)(const void*)src;
  const unsigned pad_int = (unsigned)(31 - __builtin_clz((unsigned)K) - 2); // log2(K)-2
  u32x4 g0;
  g0[0] = 1u;                                              // count=1, user-mode D#
  g0[1] = lds_off;                                         // lds_addr (bytes)
  g0[2] = (unsigned)(ga & 0xffffffffu);                    // global_addr[31:0]
  g0[3] = (unsigned)((ga >> 32) & 0x01ffffffu) | 0x80000000u; // addr[56:32] | type=2
  i32x8 g1;
  g1[0] = (int)((1u << 16)            // data_size = 2 bytes
              | (1u << 20)            // pad_enable
              | (pad_int << 22)       // pad after one row of DWORDs
              | (3u << 25));          // pad_amount = 4 DWORDs (8 bf16)
  g1[1] = (int)(((unsigned)K & 0xffffu) << 16);            // tensor_dim0[15:0]
  g1[2] = (int)((((unsigned)K >> 16) & 0xffffu) | (64u << 16)); // dim0 hi | tensor_dim1=64
  g1[3] = (int)(((unsigned)K & 0xffffu) << 16);            // dim1 hi=0 | tile_dim0=K
  g1[4] = 64;                                              // tile_dim1=64, tile_dim2=0
  g1[5] = (int)(unsigned)ld;                               // tensor_dim0_stride[31:0]
  g1[6] = 0;                                               // stride0 hi | stride1 lo
  g1[7] = 0;
  i32x4 z4 = {0, 0, 0, 0};
  i32x8 z8 = {0, 0, 0, 0, 0, 0, 0, 0};
  __builtin_amdgcn_tensor_load_to_lds(g0, g1, z4, z4, z8, 0);
}
#endif

__device__ __forceinline__ void stage_panel_fallback(const bf16* src, int ld,
                                                     int K, bf16* dst) {
  const int cpr = K >> 3;                  // 16B chunks per row
  for (int c = threadIdx.x; c < 64 * cpr; c += NTHR) {
    const int rr = c / cpr, cc = c - rr * cpr;
    *(uint4*)(dst + rr * (K + PADE) + (cc << 3)) =
        *(const uint4*)(src + (size_t)rr * ld + (cc << 3));
  }
}

// Stage A panel (64 x Ka from Asrc) and B panel (64 x Kb of weight rows) into
// LDS. One TDM-issuing wave; barrier makes the tiles visible to all 8 waves.
__device__ __forceinline__ void stage_tile(const bf16* Asrc, int lda, int Ka,
                                           const bf16* Bsrc, int ldb, int Kb,
                                           bf16* As, bf16* Bs) {
  __syncthreads();                         // previous tile's LDS readers done
#if HAS_TDM
  if (threadIdx.x < 32) {
    tdm_load_panel(Asrc, lda, lds_offset_of(As), Ka);
    tdm_load_panel(Bsrc, ldb, lds_offset_of(Bs), Kb);
    __builtin_amdgcn_s_wait_tensorcnt(0);
  }
#else
  stage_panel_fallback(Asrc, lda, Ka, As);
  stage_panel_fallback(Bsrc, ldb, Kb, Bs);
#endif
  __syncthreads();
}

// ------------------------------------------------- 64x64-tile WMMA GEMM core
// Operands in LDS (row stride K+PADE). Per wave: two 16x16 f32 accumulators.
// Fragment lane mapping: lanes 0-15 -> K-low 16, lanes 16-31 -> K-high 16.
__device__ __forceinline__ void gemm64x64_lds(const bf16* As, const bf16* Bs,
                                              int K, int mw, int nw, f32x8* acc) {
  const int lane = threadIdx.x & 31;
  const int lm   = lane & 15;
  const int koff = (lane >> 4) << 4;   // 0 or 16
  const int ldp  = K + PADE;
  const bf16* ap  = As + (mw + lm) * ldp + koff;
  const bf16* bp0 = Bs + (nw + lm) * ldp + koff;
  const bf16* bp1 = Bs + (nw + 16 + lm) * ldp + koff;
  for (int kk = 0; kk < K; kk += 32) {
    bf16x16 af = *(const bf16x16*)(ap  + kk);
    bf16x16 b0 = *(const bf16x16*)(bp0 + kk);
    bf16x16 b1 = *(const bf16x16*)(bp1 + kk);
    acc[0] = __builtin_amdgcn_wmma_f32_16x16x32_bf16(false, af, false, b0,
                                                     (short)0, acc[0], false, false);
    acc[1] = __builtin_amdgcn_wmma_f32_16x16x32_bf16(false, af, false, b1,
                                                     (short)0, acc[1], false, false);
  }
}

// C/D fragment mapping: acc[j][r] at lane l -> row m0 + 8*(l>=16) + r,
//                                             col n0 + 16*j + (l&15)
#define EPILOGUE_BEGIN                                            \
  const int lane = threadIdx.x & 31;                              \
  const int lm = lane & 15;                                       \
  const int rb = (lane >> 4) * 8;                                 \
  _Pragma("unroll") for (int j = 0; j < 2; ++j)                   \
  _Pragma("unroll") for (int r = 0; r < 8; ++r) {                 \
    const int row = m0 + rb + r;                                  \
    const int col = n0 + j * 16 + lm;
#define EPILOGUE_END }

// ----------------------------------------------------------- phase: z=tanh()
__device__ __forceinline__ void phase_ode_gemm1(const bf16* abf, const bf16* W1b,
                                                const float* b1, bf16* zbf,
                                                bf16* As, bf16* Bs) {
  const int jid = blockIdx.x;                 // exactly 128 tiles
  const int tm = (jid & 15) * 64, tn = (jid >> 4) * 64;
  stage_tile(abf + (size_t)tm * H_, H_, H_, W1b + (size_t)tn * H_, H_, H_, As, Bs);
  const int wave = threadIdx.x >> 5;
  const int mw = (wave & 3) * 16, nw = (wave >> 2) * 32;
  const int m0 = tm + mw, n0 = tn + nw;
  f32x8 acc[2] = {};
  gemm64x64_lds(As, Bs, H_, mw, nw, acc);
  EPILOGUE_BEGIN
    zbf[row * H_ + col] = (bf16)tanhf(acc[j][r] + b1[col]);
  EPILOGUE_END
}

// --------------------------------------- phase: k = z@W2^T, RK4 bookkeeping
__device__ __forceinline__ void phase_ode_gemm2(int st, float hsub,
    const bf16* zbf, const bf16* W2b, const float* b2,
    float* h, float* ksum, bf16* abf, bf16* As, bf16* Bs) {
  const int jid = blockIdx.x;
  const int tm = (jid & 15) * 64, tn = (jid >> 4) * 64;
  stage_tile(zbf + (size_t)tm * H_, H_, H_, W2b + (size_t)tn * H_, H_, H_, As, Bs);
  const int wave = threadIdx.x >> 5;
  const int mw = (wave & 3) * 16, nw = (wave >> 2) * 32;
  const int m0 = tm + mw, n0 = tn + nw;
  f32x8 acc[2] = {};
  gemm64x64_lds(As, Bs, H_, mw, nw, acc);
  const float wgt   = (st == 0 || st == 3) ? 1.0f : 2.0f;  // 1,2,2,1
  const float cnext = (st == 2) ? 1.0f : 0.5f;             // a-coef of next stage
  EPILOGUE_BEGIN
    const int idx = row * H_ + col;
    const float k  = acc[j][r] + b2[col];
    const float ks = (st == 0) ? wgt * k : ksum[idx] + wgt * k;
    if (st < 3) {
      ksum[idx] = ks;
      abf[idx]  = (bf16)(h[idx] + cnext * hsub * k);   // input of next stage
    } else {
      const float hn = h[idx] + (hsub * (1.0f / 6.0f)) * ks;
      h[idx]   = hn;
      abf[idx] = (bf16)hn;                             // input of next substep / GRU
    }
  EPILOGUE_END
}

// ------------------------------------------- phase: GRU gate GEMMs (gh & gi)
__device__ __forceinline__ void phase_gru_gemm(const bf16* abf, const bf16* xbf,
    const bf16* Whhb, const bf16* Wihb, float* gh, float* gi,
    bf16* As, bf16* Bs) {
  const int wave = threadIdx.x >> 5;
  for (int jid = blockIdx.x; jid < 768; jid += NBLK) {   // 384 gh + 384 gi tiles
    const int which = (jid >= 384);
    const int id = jid & 383;
    const int tm = (id & 15) * 64, tn = (id >> 4) * 64;  // N = 1536
    const int K = which ? I_ : H_;
    if (!which) stage_tile(abf + (size_t)tm * H_, H_, H_,
                           Whhb + (size_t)tn * H_, H_, H_, As, Bs);
    else        stage_tile(xbf + (size_t)tm * I_, I_, I_,
                           Wihb + (size_t)tn * I_, I_, I_, As, Bs);
    const int mw = (wave & 3) * 16, nw = (wave >> 2) * 32;
    const int m0 = tm + mw, n0 = tn + nw;
    f32x8 acc[2] = {};
    gemm64x64_lds(As, Bs, K, mw, nw, acc);
    float* C = which ? gi : gh;
    EPILOGUE_BEGIN
      C[row * H3_ + col] = acc[j][r];
    EPILOGUE_END
  }
}

// -------------------------------------- phase: GRU elementwise + next-x prep
__device__ __forceinline__ void phase_gru_elem(int t, const float* x,
    const float* bih, const float* bhh, const float* gi, const float* gh,
    float* h, bf16* abf, bf16* xbf) {
  const int i0 = blockIdx.x * NTHR + threadIdx.x;
  for (int idx = i0; idx < B_ * H_; idx += NBLK * NTHR) {
    const int bb = idx >> 9, jj = idx & (H_ - 1);
    const int base = bb * H3_;
    const float gr  = gi[base + jj]           + bih[jj]           + gh[base + jj]           + bhh[jj];
    const float gz  = gi[base + H_ + jj]      + bih[H_ + jj]      + gh[base + H_ + jj]      + bhh[H_ + jj];
    const float gin = gi[base + 2 * H_ + jj]  + bih[2 * H_ + jj];
    const float ghn = gh[base + 2 * H_ + jj]  + bhh[2 * H_ + jj];
    const float r = 1.0f / (1.0f + __expf(-gr));
    const float z = 1.0f / (1.0f + __expf(-gz));
    const float n = tanhf(gin + r * ghn);
    const float hn = (1.0f - z) * n + z * h[idx];
    h[idx]   = hn;
    abf[idx] = (bf16)hn;
  }
  if (t + 1 < T_) {                       // stage next observation in bf16
    for (int k = i0; k < B_ * I_; k += NBLK * NTHR) {
      const int bb = k >> 7, c = k & (I_ - 1);
      xbf[k] = (bf16)x[(size_t)bb * T_ * I_ + (size_t)(t + 1) * I_ + c];
    }
  }
}

// ------------------------------------------------------ phase: mu / logvar
__device__ __forceinline__ void phase_head(const bf16* abf, const bf16* Wmub,
    const bf16* Wlvb, const float* bmu, const float* blv, float* out,
    bf16* As, bf16* Bs) {
  const int wave = threadIdx.x >> 5;
  for (int jid = blockIdx.x; jid < 32; jid += NBLK) {    // 16 mu + 16 lv tiles
    const int which = (jid >= 16);
    const bf16* W = which ? Wlvb : Wmub;
    const float* bias = which ? blv : bmu;
    float* o = out + (size_t)which * B_ * L_;
    const int tm = (jid & 15) * 64;
    stage_tile(abf + (size_t)tm * H_, H_, H_, W, H_, H_, As, Bs);
    const int mw = (wave & 3) * 16, nw = (wave >> 2) * 32;
    const int m0 = tm + mw, n0 = nw;                     // N = 64
    f32x8 acc[2] = {};
    gemm64x64_lds(As, Bs, H_, mw, nw, acc);
    EPILOGUE_BEGIN
      o[row * L_ + col] = acc[j][r] + bias[col];
    EPILOGUE_END
  }
}

// ---------------------------------------------------------------- kernels
__global__ void __launch_bounds__(NTHR)
odegru_prep(const float* W1, const float* W2, const float* Whh, const float* Wih,
            const float* Wmu, const float* Wlv, const float* x,
            bf16* W1b, bf16* W2b, bf16* Whhb, bf16* Wihb, bf16* Wmub, bf16* Wlvb,
            float* h, bf16* abf, bf16* xbf, unsigned* bar) {
  const int i = blockIdx.x * blockDim.x + threadIdx.x;
  const int s = gridDim.x * blockDim.x;
  for (int k = i; k < H_ * H_; k += s) { W1b[k] = (bf16)W1[k]; W2b[k] = (bf16)W2[k]; }
  for (int k = i; k < H3_ * H_; k += s) Whhb[k] = (bf16)Whh[k];
  for (int k = i; k < H3_ * I_; k += s) Wihb[k] = (bf16)Wih[k];
  for (int k = i; k < L_ * H_;  k += s) { Wmub[k] = (bf16)Wmu[k]; Wlvb[k] = (bf16)Wlv[k]; }
  for (int k = i; k < B_ * H_;  k += s) { h[k] = 0.0f; abf[k] = (bf16)0.0f; }
  for (int k = i; k < B_ * I_;  k += s) {
    const int bb = k >> 7, c = k & (I_ - 1);
    xbf[k] = (bf16)x[(size_t)bb * T_ * I_ + c];          // x[b, 0, c]
  }
  if (i == 0) { bar[0] = 0; bar[1] = 0; }
}

__global__ void __launch_bounds__(NTHR)
odegru_persistent(const float* x, const float* tarr,
                  const float* bih, const float* bhh,
                  const float* b1, const float* b2,
                  const float* bmu, const float* blv,
                  const bf16* W1b, const bf16* W2b,
                  const bf16* Whhb, const bf16* Wihb,
                  const bf16* Wmub, const bf16* Wlvb,
                  float* h, float* ksum, bf16* abf, bf16* zbf,
                  float* gi, float* gh, bf16* xbf,
                  unsigned* bar, float* out) {
  extern __shared__ bf16 smem[];             // [A panel | B panel], stride H_+PADE
  bf16* As = smem;
  bf16* Bs = smem + SMEM_ELEMS;

  // ---- h0 = GRU(x[:,0,:], 0) ----
  phase_gru_gemm(abf, xbf, Whhb, Wihb, gh, gi, As, Bs);   grid_sync(bar);
  phase_gru_elem(0, x, bih, bhh, gi, gh, h, abf, xbf);    grid_sync(bar);

  // ---- scan over observation times ----
  for (int t = 1; t < T_; ++t) {
    const float dt   = tarr[t] - tarr[t - 1];
    const float hsub = dt * (1.0f / NSUB);
    for (int sub = 0; sub < NSUB; ++sub) {
      #pragma unroll 1
      for (int st = 0; st < 4; ++st) {                    // RK4 stages k1..k4
        phase_ode_gemm1(abf, W1b, b1, zbf, As, Bs);       grid_sync(bar);
        phase_ode_gemm2(st, hsub, zbf, W2b, b2, h, ksum, abf, As, Bs);
        grid_sync(bar);
      }
    }
    phase_gru_gemm(abf, xbf, Whhb, Wihb, gh, gi, As, Bs); grid_sync(bar);
    phase_gru_elem(t, x, bih, bhh, gi, gh, h, abf, xbf);  grid_sync(bar);
  }

  // ---- heads ----
  phase_head(abf, Wmub, Wlvb, bmu, blv, out, As, Bs);
}

// ---------------------------------------------------------------- launcher
extern "C" void kernel_launch(void* const* d_in, const int* in_sizes, int n_in,
                              void* d_out, int out_size, void* d_ws, size_t ws_size,
                              hipStream_t stream) {
  const float* x   = (const float*)d_in[0];
  const float* t   = (const float*)d_in[1];
  const float* Wih = (const float*)d_in[2];
  const float* Whh = (const float*)d_in[3];
  const float* bih = (const float*)d_in[4];
  const float* bhh = (const float*)d_in[5];
  const float* W1  = (const float*)d_in[6];
  const float* b1  = (const float*)d_in[7];
  const float* W2  = (const float*)d_in[8];
  const float* b2  = (const float*)d_in[9];
  const float* Wmu = (const float*)d_in[10];
  const float* bmu = (const float*)d_in[11];
  const float* Wlv = (const float*)d_in[12];
  const float* blv = (const float*)d_in[13];

  // workspace carve-up (~22.5 MB total), 256B aligned
  char* ws = (char*)d_ws;
  size_t off = 0;
  auto take = [&](size_t bytes) -> void* {
    void* p = ws + off;
    off = (off + bytes + 255) & ~(size_t)255;
    return p;
  };
  unsigned* bar  = (unsigned*)take(256);
  bf16* W1b  = (bf16*)take((size_t)H_ * H_ * 2);
  bf16* W2b  = (bf16*)take((size_t)H_ * H_ * 2);
  bf16* Whhb = (bf16*)take((size_t)H3_ * H_ * 2);
  bf16* Wihb = (bf16*)take((size_t)H3_ * I_ * 2);
  bf16* Wmub = (bf16*)take((size_t)L_ * H_ * 2);
  bf16* Wlvb = (bf16*)take((size_t)L_ * H_ * 2);
  float* h    = (float*)take((size_t)B_ * H_ * 4);
  float* ksum = (float*)take((size_t)B_ * H_ * 4);
  bf16* abf   = (bf16*)take((size_t)B_ * H_ * 2);
  bf16* zbf   = (bf16*)take((size_t)B_ * H_ * 2);
  float* gi   = (float*)take((size_t)B_ * H3_ * 4);
  float* gh   = (float*)take((size_t)B_ * H3_ * 4);
  bf16* xbf   = (bf16*)take((size_t)B_ * I_ * 2);

  odegru_prep<<<512, NTHR, 0, stream>>>(W1, W2, Whh, Wih, Wmu, Wlv, x,
                                        W1b, W2b, Whhb, Wihb, Wmub, Wlvb,
                                        h, abf, xbf, bar);

  odegru_persistent<<<NBLK, NTHR, SMEM_BYTES, stream>>>(
      x, t, bih, bhh, b1, b2, bmu, blv,
      W1b, W2b, Whhb, Wihb, Wmub, Wlvb,
      h, ksum, abf, zbf, gi, gh, xbf,
      bar, (float*)d_out);
}
